// BandSplit_66425964200235
// MI455X (gfx1250) — matrix operand order repaired and checked by reading, not compiled, not run on hardware
//
#include <hip/hip_runtime.h>

// ---------------------------------------------------------------------------
// BandSplit for MI455X (gfx1250, wave32, WMMA).
// GroupNorm folded into fp32 scalar epilogue; GEMM runs on
// v_wmma_f32_16x16x32_bf16 (memory-bound problem: ~420 MB DRAM @ 23.3 TB/s).
// ---------------------------------------------------------------------------

typedef __attribute__((ext_vector_type(16))) __bf16 v16bf;
typedef __attribute__((ext_vector_type(8)))  __bf16 v8bf;
typedef __attribute__((ext_vector_type(8)))  float  v8f;

#define TDIM   2000
#define FDIM   481
#define DDIM   128
#define NBANDS 36
#define TT     128      // T-tile per workgroup
#define PAD    8        // LDS row padding (elements) -> conflict-free frag loads

// Workspace layout (floats then bf16):
//   [0..288)      mu          (8 batches x 36 bands)
//   [288..576)    rsig
//   [576..5184)   SW  = sum_c W*nb + bc   (36 x 128)
//   [5184..9792)  SG  = sum_c W*nw        (36 x 128)
//   byte 39424+:  Wg bf16, per-band [D][chp] row-major, chp padded to 32
#define MU_OFF 0
#define RS_OFF 288
#define SW_OFF 576
#define SG_OFF 5184
#define WG_BYTE_OFF 39424

struct BandInfo { int w, ch, chp, f0, lw; size_t wgoff; };

__device__ __forceinline__ BandInfo band_info(int n) {
  BandInfo bi;
  if (n < 20)      { bi.w=8;  bi.ch=32;  bi.chp=32;  bi.f0=n*8;           bi.lw=3; bi.wgoff=(size_t)n*DDIM*32; }
  else if (n < 30) { bi.w=16; bi.ch=64;  bi.chp=64;  bi.f0=160+(n-20)*16; bi.lw=4; bi.wgoff=81920 +(size_t)(n-20)*DDIM*64; }
  else if (n < 35) { bi.w=32; bi.ch=128; bi.chp=128; bi.f0=320+(n-30)*32; bi.lw=5; bi.wgoff=163840+(size_t)(n-30)*DDIM*128; }
  else             { bi.w=1;  bi.ch=4;   bi.chp=32;  bi.f0=480;           bi.lw=0; bi.wgoff=245760; }
  return bi;
}

// channel c of band = (ri, cc, k): c = ri*2w + cc*w + k ; source row in x
__device__ __forceinline__ const float* chan_ptr(const float* xr, const float* xi,
                                                 int b, const BandInfo& bi, int c) {
  int ri  = c >> (bi.lw + 1);
  int rem = c & ((bi.w << 1) - 1);
  int cc  = rem >> bi.lw;
  int k   = rem & (bi.w - 1);
  const float* base = ri ? xi : xr;
  return base + (((size_t)(b * 2 + cc) * FDIM + bi.f0 + k) * TDIM);
}

__device__ __forceinline__ __bf16 f2bf(float f) {            // RNE truncate
  unsigned u = __builtin_bit_cast(unsigned, f);
  unsigned short h = (unsigned short)((u + 0x7FFFu + ((u >> 16) & 1u)) >> 16);
  return __builtin_bit_cast(__bf16, h);
}

// ------------------------- pass 1: GroupNorm stats -------------------------
__global__ __launch_bounds__(256)
void stats_kernel(const float* __restrict__ xr, const float* __restrict__ xi,
                  float* __restrict__ mu_out, float* __restrict__ rs_out) {
  int b = blockIdx.x / NBANDS, n = blockIdx.x % NBANDS;
  BandInfo bi = band_info(n);
  int tid = threadIdx.x;
  float s = 0.f, q = 0.f;
  for (int c = 0; c < bi.ch; ++c) {
    const float4* p = (const float4*)chan_ptr(xr, xi, b, bi, c);
    for (int i = tid; i < TDIM / 4; i += 256) {
      float4 v = p[i];
      s += v.x + v.y + v.z + v.w;
      q += v.x * v.x + v.y * v.y + v.z * v.z + v.w * v.w;
    }
  }
  __shared__ float sh_s[256], sh_q[256];
  sh_s[tid] = s; sh_q[tid] = q;
  __syncthreads();
  for (int st = 128; st > 0; st >>= 1) {
    if (tid < st) { sh_s[tid] += sh_s[tid + st]; sh_q[tid] += sh_q[tid + st]; }
    __syncthreads();
  }
  if (tid == 0) {
    float inv = 1.f / (float)(bi.ch * TDIM);
    float mu  = sh_s[0] * inv;
    float var = sh_q[0] * inv - mu * mu;
    mu_out[blockIdx.x] = mu;
    rs_out[blockIdx.x] = rsqrtf(var + 1e-5f);
  }
}

// ----------------- pass 2: fold affine into weights (bf16) -----------------
__global__ __launch_bounds__(128)
void prep_kernel(const float* nw1, const float* nb1, const float* W1, const float* bc1,
                 const float* nw2, const float* nb2, const float* W2, const float* bc2,
                 const float* nw3, const float* nb3, const float* W3, const float* bc3,
                 const float* nw4, const float* nb4, const float* W4, const float* bc4,
                 __bf16* __restrict__ wg, float* __restrict__ SW, float* __restrict__ SG) {
  int n = blockIdx.x, d = threadIdx.x;
  BandInfo bi = band_info(n);
  const float *nw, *nbv, *W, *bc; int nl;
  if (n < 20)      { nw=nw1; nbv=nb1; W=W1; bc=bc1; nl=n; }
  else if (n < 30) { nw=nw2; nbv=nb2; W=W2; bc=bc2; nl=n-20; }
  else if (n < 35) { nw=nw3; nbv=nb3; W=W3; bc=bc3; nl=n-30; }
  else             { nw=nw4; nbv=nb4; W=W4; bc=bc4; nl=0; }
  const float* Wp = W + ((size_t)nl * DDIM + d) * bi.ch;
  __bf16* wo = wg + bi.wgoff + (size_t)d * bi.chp;
  float sw = 0.f, sg = 0.f;
  for (int c = 0; c < bi.ch; ++c) {
    float wv = Wp[c], g = nw[nl * bi.ch + c];
    sw += wv * nbv[nl * bi.ch + c];
    sg += wv * g;
    wo[c] = f2bf(wv * g);
  }
  for (int c = bi.ch; c < bi.chp; ++c) wo[c] = f2bf(0.f);   // K pad (band 4)
  SW[n * DDIM + d] = sw + bc[nl * DDIM + d];
  SG[n * DDIM + d] = sg;
}

// ------------------------- pass 3: per-band WMMA GEMM ----------------------
__global__ __launch_bounds__(256)
void gemm_kernel(const float* __restrict__ xr, const float* __restrict__ xi,
                 const __bf16* __restrict__ wg,
                 const float* __restrict__ SW, const float* __restrict__ SG,
                 const float* __restrict__ muv, const float* __restrict__ rsv,
                 float* __restrict__ out) {
  __shared__ __align__(16) __bf16 As[DDIM * (128 + PAD)];   // [d][k]
  __shared__ __align__(16) __bf16 Bs[TT   * (128 + PAD)];   // [t][k] (transposed x)
  __shared__ float sb1[DDIM], sb2[DDIM];

  int n = blockIdx.y, b = blockIdx.z, t0 = blockIdx.x * TT;
  BandInfo bi = band_info(n);
  int st  = bi.chp + PAD;
  int tid = threadIdx.x;
  int ksh = (bi.chp == 32) ? 5 : (bi.chp == 64 ? 6 : 7);

  // stage A: folded weights, row-major with pad
  const __bf16* wsrc = wg + bi.wgoff;
  for (int i = tid; i < DDIM * bi.chp; i += 256)
    As[(i >> ksh) * st + (i & (bi.chp - 1))] = wsrc[i];

  // stage B: x (f32, t-contiguous) -> bf16, transposed to [t][k]
  for (int i = tid; i < bi.ch * TT; i += 256) {
    int c = i >> 7, tl = i & (TT - 1), t = t0 + tl;
    const float* p = chan_ptr(xr, xi, b, bi, c);
    Bs[tl * st + c] = f2bf((t < TDIM) ? p[t] : 0.f);
  }
  for (int i = tid; i < (bi.chp - bi.ch) * TT; i += 256)    // zero K pad
    Bs[(i & (TT - 1)) * st + bi.ch + (i >> 7)] = f2bf(0.f);

  if (tid < DDIM) { sb1[tid] = SW[n * DDIM + tid]; sb2[tid] = SG[n * DDIM + tid]; }
  __syncthreads();

  int lane = tid & 31, wv = tid >> 5;
  int wvD = wv & 3, wvT = wv >> 2;          // 4x2 wave grid over 128x128 tile
  int hi = lane >> 4, ln = lane & 15;

  v8f acc[2][4] = {};
  for (int kb = 0; kb < bi.chp; kb += 32) {
    v16bf afrag[2], bfrag[4];
#pragma unroll
    for (int i = 0; i < 2; ++i) {           // A 16x32: lane=row, K 8-blocks
      const __bf16* ap = &As[(wvD * 32 + i * 16 + ln) * st + kb + hi * 8];
      union { v16bf v; v8bf h[2]; } u;
      u.h[0] = *(const v8bf*)ap;            // K = hi*8 + 0..7
      u.h[1] = *(const v8bf*)(ap + 16);     // K = 16 + hi*8 + 0..7
      afrag[i] = u.v;
    }
#pragma unroll
    for (int j = 0; j < 4; ++j) {           // B 32x16: lane=col, K 8-blocks
      const __bf16* bp = &Bs[(wvT * 64 + j * 16 + ln) * st + kb + hi * 8];
      union { v16bf v; v8bf h[2]; } u;
      u.h[0] = *(const v8bf*)bp;
      u.h[1] = *(const v8bf*)(bp + 16);
      bfrag[j] = u.v;
    }
#pragma unroll
    for (int i = 0; i < 2; ++i)
#pragma unroll
      for (int j = 0; j < 4; ++j)
        acc[i][j] = __builtin_amdgcn_wmma_f32_16x16x32_bf16(
            false, afrag[i], false, bfrag[j], (short)0, acc[i][j], false, false);
  }

  // fp32 epilogue: out = rsig*acc + (SW - rsig*mu*SG)
  float mu = muv[b * NBANDS + n], rs = rsv[b * NBANDS + n];
  float rsmu = rs * mu;
#pragma unroll
  for (int i = 0; i < 2; ++i) {
    float bias[8];
#pragma unroll
    for (int v = 0; v < 8; ++v) {
      int d = wvD * 32 + i * 16 + hi * 8 + v;
      bias[v] = sb1[d] - rsmu * sb2[d];
    }
#pragma unroll
    for (int j = 0; j < 4; ++j) {
      int t = t0 + wvT * 64 + j * 16 + ln;
      if (t < TDIM) {
#pragma unroll
        for (int v = 0; v < 8; ++v) {
          int d = wvD * 32 + i * 16 + hi * 8 + v;
          out[(((size_t)b * DDIM + d) * NBANDS + n) * TDIM + t] =
              rs * acc[i][j][v] + bias[v];
        }
      }
    }
  }
}

// ---------------------------------------------------------------------------
extern "C" void kernel_launch(void* const* d_in, const int* in_sizes, int n_in,
                              void* d_out, int out_size, void* d_ws, size_t ws_size,
                              hipStream_t stream) {
  (void)in_sizes; (void)n_in; (void)out_size; (void)ws_size;
  const float* xr  = (const float*)d_in[0];
  const float* xi  = (const float*)d_in[1];
  const float* nw1 = (const float*)d_in[2];  const float* nb1 = (const float*)d_in[3];
  const float* W1  = (const float*)d_in[4];  const float* bc1 = (const float*)d_in[5];
  const float* nw2 = (const float*)d_in[6];  const float* nb2 = (const float*)d_in[7];
  const float* W2  = (const float*)d_in[8];  const float* bc2 = (const float*)d_in[9];
  const float* nw3 = (const float*)d_in[10]; const float* nb3 = (const float*)d_in[11];
  const float* W3  = (const float*)d_in[12]; const float* bc3 = (const float*)d_in[13];
  const float* nw4 = (const float*)d_in[14]; const float* nb4 = (const float*)d_in[15];
  const float* W4  = (const float*)d_in[16]; const float* bc4 = (const float*)d_in[17];

  float*  wsf = (float*)d_ws;
  float*  mu  = wsf + MU_OFF;
  float*  rs  = wsf + RS_OFF;
  float*  SW  = wsf + SW_OFF;
  float*  SG  = wsf + SG_OFF;
  __bf16* wgp = (__bf16*)((char*)d_ws + WG_BYTE_OFF);   // ~500 KB folded weights

  stats_kernel<<<8 * NBANDS, 256, 0, stream>>>(xr, xi, mu, rs);
  prep_kernel<<<NBANDS, 128, 0, stream>>>(nw1, nb1, W1, bc1, nw2, nb2, W2, bc2,
                                          nw3, nb3, W3, bc3, nw4, nb4, W4, bc4,
                                          wgp, SW, SG);
  gemm_kernel<<<dim3((TDIM + TT - 1) / TT, NBANDS, 8), 256, 0, stream>>>(
      xr, xi, wgp, SW, SG, mu, rs, (float*)d_out);
}
